// Set2SetNN_32993938768103
// MI455X (gfx1250) — matrix operand order repaired and checked by reading, not compiled, not run on hardware
//
#include <hip/hip_runtime.h>
#include <hip/hip_bf16.h>

// ---------------------------------------------------------------------------
// Set2Set (node + edge) + decode/FFN head for MI455X (gfx1250, wave32).
// GEMMs: V_WMMA_F32_16X16X32_BF16 with a double-buffered
// GLOBAL_LOAD_ASYNC_TO_LDS pipeline (one barrier + one asynccnt wait / K-step).
// ---------------------------------------------------------------------------

typedef __attribute__((ext_vector_type(16))) __bf16 v16bf;
typedef __attribute__((ext_vector_type(8)))  float  v8f;
typedef __attribute__((ext_vector_type(4)))  float  v4f;

static constexpr int Bg  = 2048;    // graphs
static constexpr int Nn  = 65536;   // nodes
static constexpr int Ee  = 131072;  // edges
static constexpr int DN  = 256;     // node hidden
static constexpr int DE  = 128;     // edge hidden
static constexpr int HN  = 2 * DN;  // 512
static constexpr int HE  = 2 * DE;  // 256
static constexpr int ITERS = 6;

__device__ __forceinline__ float sigmoidf_(float x) { return 1.f / (1.f + __expf(-x)); }

// Async 16B global -> LDS copy (CDNA5 async path, tracked by ASYNCcnt).
__device__ __forceinline__ void async_copy_b128(uint32_t lds_off, uint64_t gaddr) {
    asm volatile("global_load_async_to_lds_b128 %0, %1, off"
                 :: "v"(lds_off), "v"(gaddr) : "memory");
}
__device__ __forceinline__ void async_wait0() {
    asm volatile("s_wait_asynccnt 0x0" ::: "memory");
}

// ---------------------------------------------------------------------------
// Segment start offsets: start[b] = lower_bound(seg, b); seg is sorted.
// ---------------------------------------------------------------------------
__global__ void seg_starts_kernel(const int* __restrict__ seg, int M, int B,
                                  int* __restrict__ start) {
    int t = blockIdx.x * blockDim.x + threadIdx.x;
    if (t > B) return;
    int lo = 0, hi = M;
    while (lo < hi) {
        int mid = (lo + hi) >> 1;
        if (seg[mid] < t) lo = mid + 1; else hi = mid;
    }
    start[t] = lo;
}

// ---------------------------------------------------------------------------
// Pack LSTM weights into [N, K] bf16 (row n = concat(Wih[n,:], Whh[n,:]));
// bias[n] = bih[n] + bhh[n].  Coalesced along k.
// ---------------------------------------------------------------------------
__global__ void pack_lstm_w_kernel(const float* __restrict__ Wih,
                                   const float* __restrict__ Whh,
                                   const float* __restrict__ bih,
                                   const float* __restrict__ bhh,
                                   __bf16* __restrict__ Wt,
                                   float* __restrict__ bias,
                                   int N, int K1, int K2) {
    int idx = blockIdx.x * blockDim.x + threadIdx.x;
    int K = K1 + K2;
    if (idx < N) bias[idx] = bih[idx] + bhh[idx];
    if (idx >= N * K) return;
    int n = idx / K, k = idx % K;
    float v = (k < K1) ? Wih[n * K1 + k] : Whh[n * K2 + (k - K1)];
    Wt[idx] = (__bf16)v;
}

// Decode/FFN weights: f32 [K,N] -> bf16 [N,K] (transpose at pack time).
__global__ void transpose_cvt_kernel(const float* __restrict__ W,
                                     __bf16* __restrict__ Wt, int K, int N) {
    int idx = blockIdx.x * blockDim.x + threadIdx.x;
    if (idx >= N * K) return;
    int n = idx / K, k = idx % K;
    Wt[idx] = (__bf16)W[(size_t)k * N + n];
}

// out[b, 0:K1] = A1[b], out[b, K1:K1+K2] = A2[b]
__global__ void pack_concat_kernel(const float* __restrict__ A1, int K1,
                                   const float* __restrict__ A2, int K2,
                                   float* __restrict__ out) {
    int idx = blockIdx.x * blockDim.x + threadIdx.x;
    int K = K1 + K2;
    int b = idx / K, k = idx % K;
    out[idx] = (k < K1) ? A1[b * K1 + k] : A2[b * K2 + (k - K1)];
}

// ---------------------------------------------------------------------------
// GEMM: C[M,N] = A[M,K](f32) * Wt[N,K](bf16, pre-transposed) + bias[N].
// Block = 256 threads = 8 waves; block tile 128x64; wave tile 32x32
// (2x2 WMMA 16x16x32).  Double-buffered async global->LDS pipeline:
//   issue(buf0); loop { wait+barrier; issue(buf^1, kt+BK); compute(buf); }
// Requires M%128==0, N%64==0, K%32==0 (true for all call sites).
// ---------------------------------------------------------------------------
template <bool RELU>
__global__ __launch_bounds__(256)
void gemm_bf16_wmma_kernel(const float* __restrict__ A,
                           const __bf16* __restrict__ Wt,
                           const float* __restrict__ bias,
                           float* __restrict__ C,
                           int M, int N, int K) {
    constexpr int BM = 128, BN = 64, BK = 32;
    __shared__ float  As[2][BM * BK];   // [m][k] f32  (2 x 16 KB)
    __shared__ __bf16 Bs[2][BN * BK];   // [n][k] bf16 (2 x  4 KB)

    const int tid  = threadIdx.x;
    const int m0   = blockIdx.y * BM;
    const int n0   = blockIdx.x * BN;
    const int wave = tid >> 5;
    const int lane = tid & 31;
    const int wm   = (wave & 3) * 32;   // 4 waves along M
    const int wn   = (wave >> 2) * 32;  // 2 waves along N
    const int ll   = lane & 15;
    const int hi   = lane >> 4;         // lane half

    // Per-thread async-copy mappings (constant across K loop).
    //   A tile: 128 rows x 32 f32 = 1024 x 16B chunks -> 4 per thread.
    //   B tile:  64 rows x 32 bf16 = 256 x 16B chunks -> 1 per thread.
    const int bn  = tid >> 2;            // B row (n)
    const int bk8 = (tid & 3) * 8;       // B col (k), 8 bf16 = 16B

    auto issue_tile = [&](int buf, int kt) {
#pragma unroll
        for (int j = 0; j < 4; ++j) {
            int c  = tid + j * 256;
            int r  = c >> 3;             // A row (m)
            int k4 = (c & 7) * 4;        // A col (k), 4 f32 = 16B
            async_copy_b128((uint32_t)(uintptr_t)&As[buf][r * BK + k4],
                            (uint64_t)(uintptr_t)&A[(size_t)(m0 + r) * K + kt + k4]);
        }
        async_copy_b128((uint32_t)(uintptr_t)&Bs[buf][bn * BK + bk8],
                        (uint64_t)(uintptr_t)&Wt[(size_t)(n0 + bn) * K + kt + bk8]);
    };

    v8f acc[2][2] = {{{}, {}}, {{}, {}}};

    issue_tile(0, 0);
    int cur = 0;
    for (int kt = 0; kt < K; kt += BK) {
        // Own asyncs complete + barrier => whole tile visible to all waves;
        // also proves everyone is done reading the buffer we refill below.
        async_wait0();
        __syncthreads();
        if (kt + BK < K) issue_tile(cur ^ 1, kt + BK);

        // A fragments: row = wm + i*16 + ll; k-run [hi*16,+16).
        v16bf af[2];
#pragma unroll
        for (int i = 0; i < 2; ++i) {
            const float* ap = &As[cur][(wm + i * 16 + ll) * BK + hi * 16];
            v4f a0 = *(const v4f*)(ap);
            v4f a1 = *(const v4f*)(ap + 4);
            v4f a2 = *(const v4f*)(ap + 8);
            v4f a3 = *(const v4f*)(ap + 12);
            v16bf t;
#pragma unroll
            for (int e = 0; e < 4; ++e) {
                t[e]      = (__bf16)a0[e];
                t[4 + e]  = (__bf16)a1[e];
                t[8 + e]  = (__bf16)a2[e];
                t[12 + e] = (__bf16)a3[e];
            }
            af[i] = t;
        }
        // B fragments: 16 contiguous bf16 per lane (32B vector load).
        v16bf bfr[2];
#pragma unroll
        for (int j = 0; j < 2; ++j)
            bfr[j] = *(const v16bf*)&Bs[cur][(wn + j * 16 + ll) * BK + hi * 16];

#pragma unroll
        for (int i = 0; i < 2; ++i)
#pragma unroll
            for (int j = 0; j < 2; ++j)
                acc[i][j] = __builtin_amdgcn_wmma_f32_16x16x32_bf16(
                    false, af[i], false, bfr[j], (short)0, acc[i][j], false, false);
        cur ^= 1;
    }

    // Epilogue: C/D layout — lane half selects M+8, col = lane&15.
#pragma unroll
    for (int j = 0; j < 2; ++j) {
        const int col = n0 + wn + j * 16 + ll;
        const float bv = bias[col];
#pragma unroll
        for (int i = 0; i < 2; ++i) {
#pragma unroll
            for (int v = 0; v < 8; ++v) {
                int row = m0 + wm + i * 16 + v + hi * 8;
                float r = acc[i][j][v] + bv;
                if (RELU) r = fmaxf(r, 0.f);
                C[(size_t)row * N + col] = r;
            }
        }
    }
}

// ---------------------------------------------------------------------------
// LSTM cell update (torch gate order i,f,g,o): elementwise over B*D.
// ---------------------------------------------------------------------------
__global__ void lstm_update_kernel(const float* __restrict__ gates,
                                   float* __restrict__ c, float* __restrict__ h,
                                   int D) {
    int idx = blockIdx.x * blockDim.x + threadIdx.x;  // b*D + d
    int b = idx / D, d = idx % D;
    size_t base = (size_t)b * 4 * D;
    float ig = gates[base + d];
    float fg = gates[base + D + d];
    float gg = gates[base + 2 * D + d];
    float og = gates[base + 3 * D + d];
    float cn = sigmoidf_(fg) * c[idx] + sigmoidf_(ig) * tanhf(gg);
    c[idx] = cn;
    h[idx] = sigmoidf_(og) * tanhf(cn);
}

// ---------------------------------------------------------------------------
// Segment attention + readout; one 256-thread block per graph.
// q_star[b] = [ h[b] | softmax-weighted readout ].
// ---------------------------------------------------------------------------
template <int D>
__global__ __launch_bounds__(256)
void attention_kernel(const float* __restrict__ feat,
                      const int* __restrict__ start,
                      const float* __restrict__ h,
                      float* __restrict__ q_star,
                      float* __restrict__ sc) {
    __shared__ float hq[D];
    __shared__ float red[8];
    __shared__ float m_sh, den_sh;

    const int b = blockIdx.x;
    const int s = start[b], e = start[b + 1];
    const int tid = threadIdx.x, lane = tid & 31, wave = tid >> 5;

    for (int d = tid; d < D; d += 256) hq[d] = h[(size_t)b * D + d];
    if (tid == 0) den_sh = 0.f;
    __syncthreads();

    // Phase 1: scores (wave per node, float4 loads) + block max.
    float lmax = -3.4e38f;
    for (int i = s + wave; i < e; i += 8) {
        float p = 0.f;
        for (int d0 = lane * 4; d0 < D; d0 += 128) {
            v4f f = *(const v4f*)&feat[(size_t)i * D + d0];
            v4f q = *(const v4f*)&hq[d0];
            p += f[0] * q[0] + f[1] * q[1] + f[2] * q[2] + f[3] * q[3];
        }
#pragma unroll
        for (int o = 16; o > 0; o >>= 1) p += __shfl_xor(p, o, 32);
        if (lane == 0) sc[i] = p;
        lmax = fmaxf(lmax, p);
    }
    if (lane == 0) red[wave] = lmax;
    __threadfence_block();
    __syncthreads();
    if (tid == 0) {
        float m = red[0];
        for (int w = 1; w < 8; ++w) m = fmaxf(m, red[w]);
        m_sh = m;
    }
    __syncthreads();
    const float m = m_sh;

    // Phase 2a: exp in place + denominator (LDS float atomic: ds_add_f32).
    float dloc = 0.f;
    for (int i = s + tid; i < e; i += 256) {
        float ex = __expf(sc[i] - m);
        sc[i] = ex;
        dloc += ex;
    }
#pragma unroll
    for (int o = 16; o > 0; o >>= 1) dloc += __shfl_xor(dloc, o, 32);
    if (lane == 0 && dloc != 0.f) atomicAdd(&den_sh, dloc);
    __threadfence_block();
    __syncthreads();
    const float inv = den_sh > 0.f ? 1.f / den_sh : 0.f;

    // Phase 2b: readout, coalesced over feature dim.
    for (int d = tid; d < D; d += 256) {
        float acc = 0.f;
        for (int i = s; i < e; ++i) acc += sc[i] * feat[(size_t)i * D + d];
        q_star[(size_t)b * 2 * D + d]     = hq[d];
        q_star[(size_t)b * 2 * D + D + d] = acc * inv;
    }
}

// ---------------------------------------------------------------------------
// Final: pred[b] = dot(x[b,0:512], Wf3) + bf3.  One wave per row.
// ---------------------------------------------------------------------------
__global__ __launch_bounds__(256)
void final_dot_kernel(const float* __restrict__ x, const float* __restrict__ w,
                      const float* __restrict__ b3, float* __restrict__ out) {
    int tid = threadIdx.x, lane = tid & 31, wave = tid >> 5;
    int b = blockIdx.x * 8 + wave;
    float p = 0.f;
    for (int d0 = lane * 4; d0 < HN; d0 += 128) {
        v4f xv = *(const v4f*)&x[(size_t)b * HN + d0];
        v4f wv = *(const v4f*)&w[d0];
        p += xv[0] * wv[0] + xv[1] * wv[1] + xv[2] * wv[2] + xv[3] * wv[3];
    }
#pragma unroll
    for (int o = 16; o > 0; o >>= 1) p += __shfl_xor(p, o, 32);
    if (lane == 0) out[b] = p + b3[0];
}

// ---------------------------------------------------------------------------
// Host orchestration.
// ---------------------------------------------------------------------------
extern "C" void kernel_launch(void* const* d_in, const int* in_sizes, int n_in,
                              void* d_out, int out_size, void* d_ws, size_t ws_size,
                              hipStream_t stream) {
    const float* node_feat = (const float*)d_in[0];
    const float* edge_feat = (const float*)d_in[1];
    const int*   node_seg  = (const int*)d_in[2];
    const int*   edge_seg  = (const int*)d_in[3];
    const float* Wih_n = (const float*)d_in[4];
    const float* Whh_n = (const float*)d_in[5];
    const float* bih_n = (const float*)d_in[6];
    const float* bhh_n = (const float*)d_in[7];
    const float* Wih_e = (const float*)d_in[8];
    const float* Whh_e = (const float*)d_in[9];
    const float* bih_e = (const float*)d_in[10];
    const float* bhh_e = (const float*)d_in[11];
    const float* Wd1 = (const float*)d_in[12];
    const float* bd1 = (const float*)d_in[13];
    const float* Wd2 = (const float*)d_in[14];
    const float* bd2 = (const float*)d_in[15];
    const float* Wf1 = (const float*)d_in[16];
    const float* bf1 = (const float*)d_in[17];
    const float* Wf2 = (const float*)d_in[18];
    const float* bf2 = (const float*)d_in[19];
    const float* Wf3 = (const float*)d_in[20];
    const float* bf3 = (const float*)d_in[21];
    float* out = (float*)d_out;

    // Bump allocator over workspace.
    char* p = (char*)d_ws;
    auto alloc = [&](size_t bytes) -> void* {
        void* r = (void*)p;
        p += (bytes + 255) & ~(size_t)255;
        return r;
    };

    const int KN = HN + DN;  // 768  node lstm K
    const int KE = HE + DE;  // 384  edge lstm K
    const int QC = HN + HE;  // 768  decode K

    __bf16* wn_t  = (__bf16*)alloc((size_t)4 * DN * KN * 2);  // [1024, 768]
    __bf16* we_t  = (__bf16*)alloc((size_t)4 * DE * KE * 2);  // [512, 384]
    __bf16* wd1t  = (__bf16*)alloc((size_t)HN * QC * 2);      // [512, 768]
    __bf16* wd2t  = (__bf16*)alloc((size_t)4 * HN * HN * 2);  // [2048, 512]
    __bf16* wf1t  = (__bf16*)alloc((size_t)HN * 4 * HN * 2);  // [512, 2048]
    __bf16* wf2t  = (__bf16*)alloc((size_t)HN * HN * 2);      // [512, 512]
    float* bias_n = (float*)alloc(4 * DN * 4);
    float* bias_e = (float*)alloc(4 * DE * 4);

    float* h_n  = (float*)alloc((size_t)Bg * DN * 4);
    float* c_n  = (float*)alloc((size_t)Bg * DN * 4);
    float* qs_n = (float*)alloc((size_t)Bg * HN * 4);
    float* xc_n = (float*)alloc((size_t)Bg * KN * 4);
    float* g_n  = (float*)alloc((size_t)Bg * 4 * DN * 4);
    float* h_e  = (float*)alloc((size_t)Bg * DE * 4);
    float* c_e  = (float*)alloc((size_t)Bg * DE * 4);
    float* qs_e = (float*)alloc((size_t)Bg * HE * 4);
    float* xc_e = (float*)alloc((size_t)Bg * KE * 4);
    float* g_e  = (float*)alloc((size_t)Bg * 4 * DE * 4);
    float* sc_n = (float*)alloc((size_t)Nn * 4);
    float* sc_e = (float*)alloc((size_t)Ee * 4);
    int* st_n   = (int*)alloc((Bg + 1) * 4);
    int* st_e   = (int*)alloc((Bg + 1) * 4);
    float* qcat = (float*)alloc((size_t)Bg * QC * 4);
    float* d1   = (float*)alloc((size_t)Bg * HN * 4);
    float* d2   = (float*)alloc((size_t)Bg * 4 * HN * 4);
    float* x1   = (float*)alloc((size_t)Bg * HN * 4);
    float* x2   = (float*)alloc((size_t)Bg * HN * 4);

    // Zero recurrent state each call (capture-safe async memsets).
    hipMemsetAsync(h_n, 0, (size_t)Bg * DN * 4, stream);
    hipMemsetAsync(c_n, 0, (size_t)Bg * DN * 4, stream);
    hipMemsetAsync(qs_n, 0, (size_t)Bg * HN * 4, stream);
    hipMemsetAsync(h_e, 0, (size_t)Bg * DE * 4, stream);
    hipMemsetAsync(c_e, 0, (size_t)Bg * DE * 4, stream);
    hipMemsetAsync(qs_e, 0, (size_t)Bg * HE * 4, stream);

    // Segment offsets.
    seg_starts_kernel<<<(Bg + 256) / 256, 256, 0, stream>>>(node_seg, Nn, Bg, st_n);
    seg_starts_kernel<<<(Bg + 256) / 256, 256, 0, stream>>>(edge_seg, Ee, Bg, st_e);

    // Weight packing into [N,K] bf16 (once per launch).
    pack_lstm_w_kernel<<<(4 * DN * KN + 255) / 256, 256, 0, stream>>>(
        Wih_n, Whh_n, bih_n, bhh_n, wn_t, bias_n, 4 * DN, HN, DN);
    pack_lstm_w_kernel<<<(4 * DE * KE + 255) / 256, 256, 0, stream>>>(
        Wih_e, Whh_e, bih_e, bhh_e, we_t, bias_e, 4 * DE, HE, DE);
    transpose_cvt_kernel<<<(HN * QC + 255) / 256, 256, 0, stream>>>(Wd1, wd1t, QC, HN);
    transpose_cvt_kernel<<<(4 * HN * HN + 255) / 256, 256, 0, stream>>>(Wd2, wd2t, HN, 4 * HN);
    transpose_cvt_kernel<<<(HN * 4 * HN + 255) / 256, 256, 0, stream>>>(Wf1, wf1t, 4 * HN, HN);
    transpose_cvt_kernel<<<(HN * HN + 255) / 256, 256, 0, stream>>>(Wf2, wf2t, HN, HN);

    // Set2Set iterations.
    for (int it = 0; it < ITERS; ++it) {
        // ---- nodes ----
        pack_concat_kernel<<<(Bg * KN) / 256, 256, 0, stream>>>(qs_n, HN, h_n, DN, xc_n);
        gemm_bf16_wmma_kernel<false><<<dim3(4 * DN / 64, Bg / 128), 256, 0, stream>>>(
            xc_n, wn_t, bias_n, g_n, Bg, 4 * DN, KN);
        lstm_update_kernel<<<(Bg * DN) / 256, 256, 0, stream>>>(g_n, c_n, h_n, DN);
        attention_kernel<DN><<<Bg, 256, 0, stream>>>(node_feat, st_n, h_n, qs_n, sc_n);
        // ---- edges ----
        pack_concat_kernel<<<(Bg * KE) / 256, 256, 0, stream>>>(qs_e, HE, h_e, DE, xc_e);
        gemm_bf16_wmma_kernel<false><<<dim3(4 * DE / 64, Bg / 128), 256, 0, stream>>>(
            xc_e, we_t, bias_e, g_e, Bg, 4 * DE, KE);
        lstm_update_kernel<<<(Bg * DE) / 256, 256, 0, stream>>>(g_e, c_e, h_e, DE);
        attention_kernel<DE><<<Bg, 256, 0, stream>>>(edge_feat, st_e, h_e, qs_e, sc_e);
    }

    // Decode + regression head.
    pack_concat_kernel<<<(Bg * QC) / 256, 256, 0, stream>>>(qs_n, HN, qs_e, HE, qcat);
    gemm_bf16_wmma_kernel<true><<<dim3(HN / 64, Bg / 128), 256, 0, stream>>>(
        qcat, wd1t, bd1, d1, Bg, HN, QC);
    gemm_bf16_wmma_kernel<false><<<dim3(4 * HN / 64, Bg / 128), 256, 0, stream>>>(
        d1, wd2t, bd2, d2, Bg, 4 * HN, HN);
    gemm_bf16_wmma_kernel<true><<<dim3(HN / 64, Bg / 128), 256, 0, stream>>>(
        d2, wf1t, bf1, x1, Bg, HN, 4 * HN);
    gemm_bf16_wmma_kernel<true><<<dim3(HN / 64, Bg / 128), 256, 0, stream>>>(
        x1, wf2t, bf2, x2, Bg, HN, HN);
    final_dot_kernel<<<Bg / 8, 256, 0, stream>>>(x2, Wf3, bf3, out);
}
